// BloomAttention_10685878633223
// MI455X (gfx1250) — matrix operand order repaired and checked by reading, not compiled, not run on hardware
//
#include <hip/hip_runtime.h>
#include <hip/hip_bf16.h>

typedef __bf16 bf16_t;
typedef __attribute__((ext_vector_type(16))) __bf16 v16bf;
typedef __attribute__((ext_vector_type(8)))  __bf16 v8bf;
typedef __attribute__((ext_vector_type(8)))  float  v8f;
typedef __attribute__((ext_vector_type(4)))  unsigned v4u;
typedef __attribute__((ext_vector_type(4)))  int   v4i;
typedef __attribute__((ext_vector_type(8)))  int   v8i;

#define DEVFN static __device__ __forceinline__

constexpr int B_  = 2;
constexpr int S_  = 2048;
constexpr int H_  = 2048;
constexpr int NH_ = 16;
constexpr int HD_ = 128;
constexpr int M_  = B_ * S_;
constexpr float QSCALE = 0.08838834764831845f;  // 1/(sqrt(HD)*layer)

// ---------------- helpers ----------------

DEVFN bf16_t f2bf(float f) {
  union { float f; unsigned u; } v; v.f = f;
  unsigned r = v.u + 0x7FFFu + ((v.u >> 16) & 1u);   // RNE
  unsigned short s = (unsigned short)(r >> 16);
  return __builtin_bit_cast(bf16_t, s);
}

// A-fragment: 16x32 bf16, row-major source, leading dim ld (elements).
// lanes 0-15: M=lane, K={0..7,16..23}; lanes 16-31: M=lane-16, K={8..15,24..31}
DEVFN v16bf load_fragA(const bf16_t* p, int ld) {
  int lane = (int)(threadIdx.x & 31u);
  int row = lane & 15, kh = lane >> 4;
  const bf16_t* r0 = p + row * ld + kh * 8;
  v8bf lo = *(const v8bf*)(r0);
  v8bf hi = *(const v8bf*)(r0 + 16);
  return __builtin_shufflevector(lo, hi, 0,1,2,3,4,5,6,7,8,9,10,11,12,13,14,15);
}

// B-fragment: K x N (K=32,N=16) supplied as Bt = N x K row-major with ld.
// lanes 0-15: N=lane, K={0..7,16..23}; lanes 16-31: N=lane-16, K={8..15,24..31}
DEVFN v16bf load_fragB(const bf16_t* p, int ld) {
  int lane = (int)(threadIdx.x & 31u);
  int n = lane & 15, kh = lane >> 4;
  const bf16_t* r0 = p + n * ld + kh * 8;
  v8bf lo = *(const v8bf*)(r0);
  v8bf hi = *(const v8bf*)(r0 + 16);
  return __builtin_shufflevector(lo, hi, 0,1,2,3,4,5,6,7,8,9,10,11,12,13,14,15);
}

DEVFN v8f wmma_bf16(v16bf a, v16bf b, v8f c) {
  return __builtin_amdgcn_wmma_f32_16x16x32_bf16(false, a, false, b, (short)0, c,
                                                 false, false);
}

DEVFN v8f vzero8() { v8f z = {0.f,0.f,0.f,0.f,0.f,0.f,0.f,0.f}; return z; }

DEVFN unsigned lds_off_of(const void* p) {
  // generic shared pointer: low 32 bits carry the addrspace(3) byte offset
  return (unsigned)(unsigned long long)p;
}

// Issue one 2D TDM tile load: global (row-major, stride0 elems of bf16) -> LDS.
// LDS padding inserts (pad_amount+1) DWORDs after every 2^(pad_interval+1)
// DWORDs stored (bank-conflict-free fragment reads).
DEVFN void tdm_load_2d(const bf16_t* gsrc, unsigned lds_byte_off,
                       unsigned tile_d0, unsigned tile_d1,
                       unsigned tensor_d0, unsigned tensor_d1,
                       unsigned stride0,
                       unsigned pad_interval, unsigned pad_amount) {
  unsigned long long ga = (unsigned long long)(const void*)gsrc;
  v4u g0;
  g0[0] = 1u;                                   // count=1, user descriptor
  g0[1] = lds_byte_off;                         // lds_addr
  g0[2] = (unsigned)ga;                         // global_addr[31:0]
  g0[3] = (unsigned)((ga >> 32) & 0x01FFFFFFull) | (2u << 30);  // addr[56:32] | type=2
  v8i g1;
  g1[0] = (int)((1u << 16) |                    // data_size = 1 -> 2 bytes
                (1u << 20) |                    // pad_enable
                (pad_interval << 22) | (pad_amount << 25));
  g1[1] = (int)((tensor_d0 & 0xFFFFu) << 16);                    // tensor_dim0[15:0]
  g1[2] = (int)((tensor_d0 >> 16) | ((tensor_d1 & 0xFFFFu) << 16));
  g1[3] = (int)((tensor_d1 >> 16) | (tile_d0 << 16));            // tile_dim0
  g1[4] = (int)tile_d1;                                          // tile_dim1 (dim2=0)
  g1[5] = (int)stride0;                                          // dim0 stride lo
  g1[6] = 0;                                                     // stride hi / dim1 stride
  g1[7] = 0;
  v4i z4 = {0, 0, 0, 0};
#if defined(__clang_major__) && (__clang_major__ >= 23)
  v8i z8 = {0, 0, 0, 0, 0, 0, 0, 0};
  __builtin_amdgcn_tensor_load_to_lds(g0, g1, z4, z4, z8, 0);
#else
  __builtin_amdgcn_tensor_load_to_lds(g0, g1, z4, z4, 0);
#endif
}

// ---------------- kernels ----------------

__global__ __launch_bounds__(256) void cvt_f32_bf16(const float* __restrict__ in,
                                                    bf16_t* __restrict__ out, int n) {
  int i = blockIdx.x * blockDim.x + threadIdx.x;
  if (i < n) out[i] = f2bf(in[i]);
}

// GEMM tiling: block = 8 waves, 256 rows x 64 cols; each wave owns 32x64
// (2 A-frags x 4 B-frags = 8 WMMAs per 32-deep k-step; B-frags read once
// from the TDM-staged LDS tile and reused for both M halves).
constexpr int LDW = 40;   // 32 elems + 8 pad (64B row + 16B pad)

// fused QKV GEMM + scatter into Q (prescaled), K, and V-transposed.
__global__ __launch_bounds__(256) void qkv_gemm(const bf16_t* __restrict__ A,
                                                const bf16_t* __restrict__ W,
                                                const float*  __restrict__ bias,
                                                bf16_t* __restrict__ Q,
                                                bf16_t* __restrict__ Kd,
                                                bf16_t* __restrict__ VT) {
  __shared__ __attribute__((aligned(16))) bf16_t wtile[2][64][LDW];
  int wid  = (int)(threadIdx.x >> 5);
  int lane = (int)(threadIdx.x & 31u);
  int col0 = blockIdx.x * 64;
  int row0 = blockIdx.y * 256 + wid * 32;

  v8f acc[2][4];
  #pragma unroll
  for (int m = 0; m < 2; ++m)
    #pragma unroll
    for (int j = 0; j < 4; ++j) acc[m][j] = vzero8();

  if (wid == 0)
    tdm_load_2d(W + (long)col0 * H_, lds_off_of(&wtile[0][0][0]),
                32, 64, H_, 3 * H_, H_, 3, 3);

  int cur = 0;
  for (int k0 = 0; k0 < H_; k0 += 32) {
    if (wid == 0) {
      int kn = (k0 + 32 < H_) ? (k0 + 32) : 0;   // tail prefetch wraps (unread)
      tdm_load_2d(W + (long)col0 * H_ + kn, lds_off_of(&wtile[cur ^ 1][0][0]),
                  32, 64, H_, 3 * H_, H_, 3, 3);
      __builtin_amdgcn_s_wait_tensorcnt(1);
    }
    __syncthreads();
    v16bf a0 = load_fragA(A + (long)row0 * H_ + k0, H_);
    v16bf a1 = load_fragA(A + (long)(row0 + 16) * H_ + k0, H_);
    #pragma unroll
    for (int j = 0; j < 4; ++j) {
      v16bf bq = load_fragB(&wtile[cur][j * 16][0], LDW);
      acc[0][j] = wmma_bf16(a0, bq, acc[0][j]);
      acc[1][j] = wmma_bf16(a1, bq, acc[1][j]);
    }
    __syncthreads();
    cur ^= 1;
  }

  int rh = (lane >> 4) * 8;
  int cl = lane & 15;
  #pragma unroll
  for (int m = 0; m < 2; ++m) {
    #pragma unroll
    for (int j = 0; j < 4; ++j) {
      int col  = col0 + j * 16 + cl;
      int head = col / (3 * HD_);
      int w    = col % (3 * HD_);
      float bv = bias[col];
      #pragma unroll
      for (int i = 0; i < 8; ++i) {
        int row = row0 + m * 16 + rh + i;
        int b   = row / S_, s = row % S_;
        float v = acc[m][j][i] + bv;
        long bh = (long)(b * NH_ + head);
        if (w < HD_)
          Q[(bh * S_ + s) * HD_ + w] = f2bf(v * QSCALE);
        else if (w < 2 * HD_)
          Kd[(bh * S_ + s) * HD_ + (w - HD_)] = f2bf(v);
        else
          VT[(bh * HD_ + (w - 2 * HD_)) * (long)S_ + s] = f2bf(v);
      }
    }
  }
}

// flash attention: K tile (32x128) and V^T tile (128x32) staged by TDM and
// shared by all 8 waves; block-uniform causal loop bound; online softmax.
constexpr int LDK = 136;  // 128 elems + 8 pad (256B + 16B)
constexpr int LDV = 40;   // 32 elems + 8 pad  (64B + 16B)

__global__ __launch_bounds__(256) void flash_attn(const bf16_t* __restrict__ Q,
                                                  const bf16_t* __restrict__ K,
                                                  const bf16_t* __restrict__ VT,
                                                  bf16_t* __restrict__ CTX) {
  __shared__ __attribute__((aligned(16))) bf16_t ktile[2][32][LDK];   // 17408 B
  __shared__ __attribute__((aligned(16))) bf16_t vtile[2][128][LDV];  // 20480 B
  __shared__ __attribute__((aligned(16))) bf16_t ldsP[8][16][32];     //  8192 B

  int wid  = (int)(threadIdx.x >> 5);
  int lane = (int)(threadIdx.x & 31u);
  int head = blockIdx.y;
  int b    = blockIdx.z;
  int qbase = (blockIdx.x * 8 + wid) * 16;
  long bh  = (long)(b * NH_ + head);

  const bf16_t* Qh = Q  + bh * (long)S_ * HD_;
  const bf16_t* Kh = K  + bh * (long)S_ * HD_;
  const bf16_t* Vh = VT + bh * (long)HD_ * S_;
  float slope = exp2f(-0.5f * (float)(head + 1));  // BLOOM ALiBi, NH=16

  v16bf aq[4];
  #pragma unroll
  for (int d = 0; d < 4; ++d)
    aq[d] = load_fragA(Qh + (long)qbase * HD_ + d * 32, HD_);

  v8f o[8];
  #pragma unroll
  for (int c = 0; c < 8; ++c) o[c] = vzero8();
  float mrun[8], lrun[8];
  #pragma unroll
  for (int i = 0; i < 8; ++i) { mrun[i] = -1e30f; lrun[i] = 0.f; }

  int rh = (lane >> 4) * 8;
  int cl = lane & 15;
  int kend_blk = blockIdx.x * 128 + 127;   // block-uniform causal bound

  if (wid == 0) {
    tdm_load_2d(Kh, lds_off_of(&ktile[0][0][0]), HD_, 32, HD_, S_, HD_, 5, 3);
    tdm_load_2d(Vh, lds_off_of(&vtile[0][0][0]), 32, HD_, S_, HD_, S_, 3, 3);
  }

  int cur = 0;
  for (int k0 = 0; k0 <= kend_blk; k0 += 32) {
    if (wid == 0) {
      int kn = (k0 + 32 <= kend_blk) ? (k0 + 32) : k0;  // tail prefetch repeats
      tdm_load_2d(Kh + (long)kn * HD_, lds_off_of(&ktile[cur ^ 1][0][0]),
                  HD_, 32, HD_, S_, HD_, 5, 3);
      tdm_load_2d(Vh + kn, lds_off_of(&vtile[cur ^ 1][0][0]),
                  32, HD_, S_, HD_, S_, 3, 3);
      __builtin_amdgcn_s_wait_tensorcnt(2);
    }
    __syncthreads();

    v8f s0 = vzero8(), s1 = vzero8();
    #pragma unroll
    for (int d = 0; d < 4; ++d) {
      s0 = wmma_bf16(aq[d], load_fragB(&ktile[cur][0][d * 32],  LDK), s0);
      s1 = wmma_bf16(aq[d], load_fragB(&ktile[cur][16][d * 32], LDK), s1);
    }

    int kp0 = k0 + cl, kp1 = k0 + 16 + cl;
    float a0 = slope * (float)kp0;
    float a1 = slope * (float)kp1;

    #pragma unroll
    for (int i = 0; i < 8; ++i) {
      int qrow = qbase + rh + i;
      float x0 = (kp0 <= qrow) ? (s0[i] + a0) : -1e30f;
      float x1 = (kp1 <= qrow) ? (s1[i] + a1) : -1e30f;
      float mx = fmaxf(x0, x1);
      #pragma unroll
      for (int off = 8; off >= 1; off >>= 1)
        mx = fmaxf(mx, __shfl_xor(mx, off, 32));   // reduce within 16-lane half
      float mn = fmaxf(mrun[i], mx);
      float p0 = __expf(x0 - mn);
      float p1 = __expf(x1 - mn);
      float rs = p0 + p1;
      #pragma unroll
      for (int off = 8; off >= 1; off >>= 1)
        rs += __shfl_xor(rs, off, 32);
      float corr = __expf(mrun[i] - mn);
      lrun[i] = lrun[i] * corr + rs;
      mrun[i] = mn;
      #pragma unroll
      for (int c = 0; c < 8; ++c) o[c][i] *= corr;
      ldsP[wid][rh + i][cl]      = f2bf(p0);
      ldsP[wid][rh + i][16 + cl] = f2bf(p1);
    }

    v16bf pa = load_fragA(&ldsP[wid][0][0], 32);   // P relayout via LDS
    #pragma unroll
    for (int c = 0; c < 8; ++c)
      o[c] = wmma_bf16(pa, load_fragB(&vtile[cur][c * 16][0], LDV), o[c]);

    __syncthreads();
    cur ^= 1;
  }

  bf16_t* ctx = CTX + ((long)b * S_ + qbase) * H_ + head * HD_;
  #pragma unroll
  for (int i = 0; i < 8; ++i) {
    float inv = 1.0f / lrun[i];
    #pragma unroll
    for (int c = 0; c < 8; ++c)
      ctx[(long)(rh + i) * H_ + c * 16 + cl] = f2bf(o[c][i] * inv);
  }
}

// dense projection: out = residual + ctx @ Wdense^T + b_dense  (fp32 out)
__global__ __launch_bounds__(256) void dense_gemm(const bf16_t* __restrict__ A,
                                                  const bf16_t* __restrict__ W,
                                                  const float*  __restrict__ bias,
                                                  const float*  __restrict__ residual,
                                                  float* __restrict__ out) {
  __shared__ __attribute__((aligned(16))) bf16_t wtile[2][64][LDW];
  int wid  = (int)(threadIdx.x >> 5);
  int lane = (int)(threadIdx.x & 31u);
  int col0 = blockIdx.x * 64;
  int row0 = blockIdx.y * 256 + wid * 32;

  v8f acc[2][4];
  #pragma unroll
  for (int m = 0; m < 2; ++m)
    #pragma unroll
    for (int j = 0; j < 4; ++j) acc[m][j] = vzero8();

  if (wid == 0)
    tdm_load_2d(W + (long)col0 * H_, lds_off_of(&wtile[0][0][0]),
                32, 64, H_, H_, H_, 3, 3);

  int cur = 0;
  for (int k0 = 0; k0 < H_; k0 += 32) {
    if (wid == 0) {
      int kn = (k0 + 32 < H_) ? (k0 + 32) : 0;
      tdm_load_2d(W + (long)col0 * H_ + kn, lds_off_of(&wtile[cur ^ 1][0][0]),
                  32, 64, H_, H_, H_, 3, 3);
      __builtin_amdgcn_s_wait_tensorcnt(1);
    }
    __syncthreads();
    v16bf a0 = load_fragA(A + (long)row0 * H_ + k0, H_);
    v16bf a1 = load_fragA(A + (long)(row0 + 16) * H_ + k0, H_);
    #pragma unroll
    for (int j = 0; j < 4; ++j) {
      v16bf bw = load_fragB(&wtile[cur][j * 16][0], LDW);
      acc[0][j] = wmma_bf16(a0, bw, acc[0][j]);
      acc[1][j] = wmma_bf16(a1, bw, acc[1][j]);
    }
    __syncthreads();
    cur ^= 1;
  }

  int rh = (lane >> 4) * 8;
  int cl = lane & 15;
  #pragma unroll
  for (int m = 0; m < 2; ++m) {
    #pragma unroll
    for (int j = 0; j < 4; ++j) {
      int col  = col0 + j * 16 + cl;
      float bv = bias[col];
      #pragma unroll
      for (int i = 0; i < 8; ++i) {
        long idx = (long)(row0 + m * 16 + rh + i) * H_ + col;
        out[idx] = residual[idx] + acc[m][j][i] + bv;
      }
    }
  }
}

// ---------------- launch ----------------

extern "C" void kernel_launch(void* const* d_in, const int* in_sizes, int n_in,
                              void* d_out, int out_size, void* d_ws, size_t ws_size,
                              hipStream_t stream) {
  (void)in_sizes; (void)n_in; (void)out_size; (void)ws_size;
  const float* hidden   = (const float*)d_in[0];
  const float* residual = (const float*)d_in[1];
  // d_in[2] = attention_mask (causal, reconstructed analytically)
  // d_in[3] = alibi          (recomputed analytically)
  const float* Wqkv     = (const float*)d_in[4];
  const float* bqkv     = (const float*)d_in[5];
  const float* Wdense   = (const float*)d_in[6];
  const float* bdense   = (const float*)d_in[7];
  float* out = (float*)d_out;

  const size_t nHid  = (size_t)M_ * H_;
  const size_t nWqkv = (size_t)3 * H_ * H_;
  const size_t nWd   = (size_t)H_ * H_;
  const size_t nHead = (size_t)B_ * NH_ * S_ * HD_;

  char* ws = (char*)d_ws;
  size_t off = 0;
  auto carve = [&](size_t elems) {
    void* p = (void*)(ws + off);
    off = (off + elems * sizeof(bf16_t) + 255) & ~(size_t)255;
    return (bf16_t*)p;
  };
  bf16_t* hb   = carve(nHid);
  bf16_t* wqb  = carve(nWqkv);
  bf16_t* wdb  = carve(nWd);
  bf16_t* Qb   = carve(nHead);
  bf16_t* Kb   = carve(nHead);
  bf16_t* VTb  = carve(nHead);
  bf16_t* ctxb = carve(nHid);

  cvt_f32_bf16<<<(int)((nHid  + 255) / 256), 256, 0, stream>>>(hidden, hb,  (int)nHid);
  cvt_f32_bf16<<<(int)((nWqkv + 255) / 256), 256, 0, stream>>>(Wqkv,   wqb, (int)nWqkv);
  cvt_f32_bf16<<<(int)((nWd   + 255) / 256), 256, 0, stream>>>(Wdense, wdb, (int)nWd);

  qkv_gemm<<<dim3(6 * H_ / 64, M_ / 256), 256, 0, stream>>>(hb, wqb, bqkv, Qb, Kb, VTb);
  flash_attn<<<dim3(S_ / 128, NH_, B_), 256, 0, stream>>>(Qb, Kb, VTb, ctxb);
  dense_gemm<<<dim3(H_ / 64, M_ / 256), 256, 0, stream>>>(ctxb, wdb, bdense, residual, out);
}
